// PyRNN_77790447665791
// MI455X (gfx1250) — compile-verified, matrix operand order checked by reading, not run on hardware
//
#include <hip/hip_runtime.h>

typedef __bf16 bf16;
typedef __attribute__((ext_vector_type(16))) __bf16 v16bf;
typedef __attribute__((ext_vector_type(8)))  __bf16 v8bf;
typedef __attribute__((ext_vector_type(8)))  float  v8f;

#define HDIM  256
#define BATCH 64
#define LSEQ  2048
#define NTHR  512            // 16 waves of 32
#define ROWS_PER_WG 512      // phase-1 GEMM rows per workgroup

// Scheduling hints: group DS-reads ahead of the WMMA chain so the wait pass
// emits partial s_wait_dscnt counts instead of fencing every WMMA to zero.
#if __has_builtin(__builtin_amdgcn_sched_group_barrier)
#define SCHED_DS_READS(n)  __builtin_amdgcn_sched_group_barrier(0x100, (n), 0)
#define SCHED_WMMAS(n)     __builtin_amdgcn_sched_group_barrier(0x008, (n), 0)
#else
#define SCHED_DS_READS(n)
#define SCHED_WMMAS(n)
#endif

// Load one 16x32 (A) or 32x16 (B) bf16 WMMA fragment from a row-major
// [16][256] (A: rows) or [256][256] (B: W^T, n-major) LDS buffer.
// Lane mapping per CDNA5 ISA: lane%16 = M (or N), lane/16 selects K half;
// fragment element j covers K = c + (j<8 ? j : j+8), c = (lane/16)*8 + kb*32.
static __device__ __forceinline__ v16bf ld_frag16(const bf16* rowp, int c) {
  v8bf lo = *(const v8bf*)(rowp + c);        // K = c .. c+7   (16B LDS load)
  v8bf hi = *(const v8bf*)(rowp + c + 16);   // K = c+16..c+23 (16B LDS load)
  return __builtin_shufflevector(lo, hi, 0,1,2,3,4,5,6,7,8,9,10,11,12,13,14,15);
}

// Load this wave's 16-column slice of a [H][H] f32 weight (row-major k,n)
// into 8 register-resident bf16 B-fragments. One-time cost.
static __device__ __forceinline__ void load_w_regs(const float* __restrict__ W,
                                                   int ng, int half, v16bf* frag) {
#pragma unroll
  for (int kb = 0; kb < 8; ++kb) {
    const int c = half * 8 + kb * 32;
    v16bf f;
#pragma unroll
    for (int j = 0; j < 16; ++j) {
      const int k = c + (j < 8 ? j : j + 8);
      f[j] = (bf16)W[k * HDIM + ng];
    }
    frag[kb] = f;
  }
}

// ---------------- Phase 1: XW0 = x @ WI0 + (BI0 + BH0)  -> staged in d_out ---
__global__ __launch_bounds__(NTHR, 1)
void xw0_gemm(const float* __restrict__ x, const float* __restrict__ WI,
              const float* __restrict__ BI, const float* __restrict__ BH,
              float* __restrict__ xw) {
  __shared__ bf16 xs[16 * HDIM];                    // one 16-row A tile, bf16
  const int tid = threadIdx.x, w = tid >> 5, lane = tid & 31;
  const int nl = lane & 15, half = lane >> 4;
  const int ng = w * 16 + nl;                       // this lane's output column

  v16bf wfr[8];
  load_w_regs(WI, ng, half, wfr);                   // WI layer 0, reg-resident
  const float bias = BI[ng] + BH[ng];               // fold both layer-0 biases
  const size_t rowbase = (size_t)blockIdx.x * ROWS_PER_WG;

  for (int mt = 0; mt < ROWS_PER_WG / 16; ++mt) {
    const size_t r0 = rowbase + (size_t)mt * 16;
    __syncthreads();
    for (int i = tid; i < 16 * HDIM; i += NTHR) {   // coalesced f32 -> bf16
      const int r = i >> 8, c = i & 255;
      xs[i] = (bf16)x[(r0 + r) * HDIM + c];
    }
    __syncthreads();

    v16bf a[8];
#pragma unroll
    for (int kb = 0; kb < 8; ++kb)
      a[kb] = ld_frag16(xs + nl * HDIM, half * 8 + kb * 32);

    v8f acc;
#pragma unroll
    for (int v = 0; v < 8; ++v) acc[v] = bias;
#pragma unroll
    for (int kb = 0; kb < 8; ++kb)
      acc = __builtin_amdgcn_wmma_f32_16x16x32_bf16(false, a[kb], false, wfr[kb],
                                                    (short)0, acc, false, false);
    SCHED_DS_READS(16);    // all 16 ds_load_b128 first ...
    SCHED_WMMAS(8);        // ... then the 8-WMMA chain (partial dscnt waits)

#pragma unroll
    for (int v = 0; v < 8; ++v) {                   // C layout: M = v + 8*half
      const int m = v + half * 8;
      xw[(r0 + m) * HDIM + ng] = acc[v];
    }
  }
}

// ---------------- Phase 2: serial recurrence, 4 WGs x 16 batch rows ----------
__global__ __launch_bounds__(NTHR, 1)
void rnn_recur(const float* __restrict__ h0in,
               const float* __restrict__ WI, const float* __restrict__ BI,
               const float* __restrict__ WH, const float* __restrict__ BH,
               float* out) {
  extern __shared__ bf16 smem[];
  bf16* WT0 = smem;                         // WH0^T [n][k] bf16, 128 KB
  bf16* WT1 = WT0 + HDIM * HDIM;            // WH1^T, 128 KB
  bf16* h0b = WT1 + HDIM * HDIM;            // [2][16][256] double-buffered
  bf16* h1b = h0b + 2 * 16 * HDIM;          // [2][16][256]

  const int tid = threadIdx.x, g = blockIdx.x;
  const int w = tid >> 5, lane = tid & 31, nl = lane & 15, half = lane >> 4;
  const int ng = w * 16 + nl;

  // One-time: transpose recurrent weights into LDS as bf16 (n-major so B
  // fragments are two contiguous 16B LDS reads, same pattern as A).
  for (int i = tid; i < HDIM * HDIM; i += NTHR) {
    const int k = i >> 8, n = i & 255;
    WT0[n * HDIM + k] = (bf16)WH[i];
    WT1[n * HDIM + k] = (bf16)WH[HDIM * HDIM + i];
  }
  // Initial hidden state into buffer 1 (step 0 reads prv = 1).
  for (int i = tid; i < 16 * HDIM; i += NTHR) {
    const int r = i >> 8, c = i & 255;
    h0b[16 * HDIM + i] = (bf16)h0in[(0 * BATCH + g * 16 + r) * HDIM + c];
    h1b[16 * HDIM + i] = (bf16)h0in[(1 * BATCH + g * 16 + r) * HDIM + c];
  }
  v16bf wi1[8];
  load_w_regs(WI + HDIM * HDIM, ng, half, wi1);   // WI layer 1, reg-resident
  const float bias1 = BI[HDIM + ng] + BH[HDIM + ng];
  const size_t hfin = (size_t)BATCH * LSEQ * HDIM;

  // XW0 tile register double-buffer: this lane's 8 accumulator-layout values.
  // Base pointer for (m = half*8, t = 0); time stride is HDIM floats.
  const float* xwp0 = out + ((size_t)(g * 16 + half * 8) * LSEQ) * HDIM + ng;
  float xw_cur[8];
#pragma unroll
  for (int v = 0; v < 8; ++v)                 // load step-0 tile up front
    xw_cur[v] = xwp0[(size_t)v * LSEQ * HDIM];

  __syncthreads();

  for (int t = 0; t < LSEQ; ++t) {
    const int cur = t & 1, prv = cur ^ 1;
    bf16* h0r = h0b + prv * 16 * HDIM;  bf16* h0w = h0b + cur * 16 * HDIM;
    bf16* h1r = h1b + prv * 16 * HDIM;  bf16* h1w = h1b + cur * 16 * HDIM;

    // ---- layer 0: acc = XW0(t) (preloaded last step) + h0_prev @ WH0 ----
    v16bf a[8], b[8];
#pragma unroll
    for (int kb = 0; kb < 8; ++kb) {
      const int c = half * 8 + kb * 32;
      a[kb] = ld_frag16(h0r + nl * HDIM, c);
      b[kb] = ld_frag16(WT0 + ng * HDIM, c);
    }
    v8f acc;
#pragma unroll
    for (int v = 0; v < 8; ++v) acc[v] = xw_cur[v];
#pragma unroll
    for (int kb = 0; kb < 8; ++kb)
      acc = __builtin_amdgcn_wmma_f32_16x16x32_bf16(false, a[kb], false, b[kb],
                                                    (short)0, acc, false, false);
    SCHED_DS_READS(32);    // 32 ds_load_b128 (a+b fragments) up front
    SCHED_WMMAS(8);        // then the 8-WMMA chain

    float r0[8];
#pragma unroll
    for (int v = 0; v < 8; ++v) {
      r0[v] = tanhf(acc[v]);
      h0w[(v + half * 8) * HDIM + ng] = (bf16)r0[v];
    }
    if (t == LSEQ - 1) {
#pragma unroll
      for (int v = 0; v < 8; ++v)
        out[hfin + (size_t)(g * 16 + v + half * 8) * HDIM + ng] = r0[v];
    }

    // Issue next step's XW0 loads now; they complete under layer-1 compute,
    // taking the global-load latency off the serial critical path. Also
    // prefetch t+2 to pre-warm L2 (XW0 staging likely L2-resident: 134 MB).
    const int tn = (t + 1 < LSEQ) ? (t + 1) : t;   // branchless clamp
    float xw_nxt[8];
#pragma unroll
    for (int v = 0; v < 8; ++v) {
      const float* p = xwp0 + (size_t)v * LSEQ * HDIM + (size_t)tn * HDIM;
      xw_nxt[v] = *p;
      __builtin_prefetch(p + HDIM, 0, 1);          // t+2, global_prefetch_b8
    }

    __syncthreads();                       // h0(t) visible to all waves

    // ---- layer 1: acc = bias1 + h0_new @ WI1 + h1_prev @ WH1 ----
#pragma unroll
    for (int kb = 0; kb < 8; ++kb)
      a[kb] = ld_frag16(h0w + nl * HDIM, half * 8 + kb * 32);
#pragma unroll
    for (int v = 0; v < 8; ++v) acc[v] = bias1;
#pragma unroll
    for (int kb = 0; kb < 8; ++kb)
      acc = __builtin_amdgcn_wmma_f32_16x16x32_bf16(false, a[kb], false, wi1[kb],
                                                    (short)0, acc, false, false);
#pragma unroll
    for (int kb = 0; kb < 8; ++kb) {
      const int c = half * 8 + kb * 32;
      a[kb] = ld_frag16(h1r + nl * HDIM, c);
      b[kb] = ld_frag16(WT1 + ng * HDIM, c);
    }
#pragma unroll
    for (int kb = 0; kb < 8; ++kb)
      acc = __builtin_amdgcn_wmma_f32_16x16x32_bf16(false, a[kb], false, b[kb],
                                                    (short)0, acc, false, false);
    SCHED_DS_READS(8);     // h0_new fragments
    SCHED_WMMAS(4);        // start WI1 chain while...
    SCHED_DS_READS(32);    // ...h1/WH1 fragments stream in
    SCHED_WMMAS(12);       // remaining WMMAs

#pragma unroll
    for (int v = 0; v < 8; ++v) {
      const float r1 = tanhf(acc[v]);
      const int m = v + half * 8;
      out[((size_t)(g * 16 + m) * LSEQ + t) * HDIM + ng] = r1;  // overwrite XW0(t)
      h1w[m * HDIM + ng] = (bf16)r1;
      if (t == LSEQ - 1)
        out[hfin + (size_t)(BATCH + g * 16 + m) * HDIM + ng] = r1;
    }
#pragma unroll
    for (int v = 0; v < 8; ++v) xw_cur[v] = xw_nxt[v];

    __syncthreads();                       // step boundary: safe to swap buffers
  }
}

extern "C" void kernel_launch(void* const* d_in, const int* in_sizes, int n_in,
                              void* d_out, int out_size, void* d_ws, size_t ws_size,
                              hipStream_t stream) {
  (void)in_sizes; (void)n_in; (void)out_size; (void)d_ws; (void)ws_size;
  const float* x  = (const float*)d_in[0];
  const float* h0 = (const float*)d_in[1];
  const float* WI = (const float*)d_in[2];
  const float* BI = (const float*)d_in[3];
  const float* WH = (const float*)d_in[4];
  const float* BH = (const float*)d_in[5];
  float* out = (float*)d_out;

  // Phase 1: parallel GEMM, stages XW0 into d_out (exactly B*L*H floats).
  const int wgs = (BATCH * LSEQ) / ROWS_PER_WG;   // 256 workgroups
  xw0_gemm<<<dim3(wgs), dim3(NTHR), 0, stream>>>(x, WI, BI, BH, out);

  // Phase 2: 4 persistent workgroups, 288 KB LDS each (CDNA5 320 KB WGP LDS).
  const size_t lds = (size_t)(2 * HDIM * HDIM + 4 * 16 * HDIM) * sizeof(bf16);
  rnn_recur<<<dim3(BATCH / 16), dim3(NTHR), lds, stream>>>(h0, WI, BI, WH, BH, out);
}
